// CrissCrossAttention_6571299963612
// MI455X (gfx1250) — matrix operand order repaired and checked by reading, not compiled, not run on hardware
//
#include <hip/hip_runtime.h>
#include <hip/hip_bf16.h>

// ---- problem constants ----
#define HD   96
#define WD   96
#define HWXY (HD*WD)        // 9216
#define CCH  512
#define CQKD 64
#define BB   8
#define NTOT (BB*HWXY)      // 73728
#define OQKV 640            // 64 q + 64 k + 512 v

typedef __attribute__((ext_vector_type(4)))  unsigned int u32x4;
typedef __attribute__((ext_vector_type(16))) __bf16       v16bf;
typedef __attribute__((ext_vector_type(8)))  float        v8f;

union Tile16 { u32x4 q[2]; v16bf v; };

__device__ __forceinline__ unsigned short f2bf(float f) {
  unsigned int u = __float_as_uint(f);
  u += 0x7FFFu + ((u >> 16) & 1u);           // round-to-nearest-even
  return (unsigned short)(u >> 16);
}

// A tile (16x32 bf16), D = A*B. Rows M strided by row_stride (elements),
// K contiguous in memory. ISA layout: lanes 0-15 hold K 0..7 & 16..23,
// lanes 16-31 hold K 8..15 & 24..31, row M = lane&15.
__device__ __forceinline__ v16bf load_a(const unsigned short* base,
                                        size_t row_stride, int lane) {
  Tile16 t;
  const unsigned short* p = base + (size_t)(lane & 15) * row_stride + ((lane >> 4) << 3);
  t.q[0] = *reinterpret_cast<const u32x4*>(p);        // K = kh*8 + 0..7
  t.q[1] = *reinterpret_cast<const u32x4*>(p + 16);   // K = 16 + kh*8 + 0..7
  return t.v;
}

// B tile (32x16 bf16). Column N strided by col_stride (elements), K contiguous.
// ISA layout: lane N = lane&15; lanes 0-15 hold K 0..7 & 16..23, lanes 16-31 K 8..15 & 24..31.
__device__ __forceinline__ v16bf load_b(const unsigned short* base,
                                        size_t col_stride, int lane) {
  Tile16 t;
  const unsigned short* p = base + (size_t)(lane & 15) * col_stride + ((lane >> 4) << 3);
  t.q[0] = *reinterpret_cast<const u32x4*>(p);        // K = kh*8 + 0..7
  t.q[1] = *reinterpret_cast<const u32x4*>(p + 16);   // K = 16 + kh*8 + 0..7
  return t.v;
}

__device__ __forceinline__ v8f wmma_bf16(v16bf a, v16bf b, v8f c) {
  // (neg_a, A, neg_b, B, c_mod, C, reuse_a, reuse_b)
  return __builtin_amdgcn_wmma_f32_16x16x32_bf16(false, a, false, b, (short)0, c, false, false);
}

// ---------------- kernel 1: pack weights to bf16 (640 x 512) + bias ----------------
__global__ void k_pack(const float* __restrict__ Wq, const float* __restrict__ bq,
                       const float* __restrict__ Wk, const float* __restrict__ bk,
                       const float* __restrict__ Wv, const float* __restrict__ bv,
                       unsigned short* __restrict__ Wqkv, float* __restrict__ bqkv) {
  int i = blockIdx.x * 256 + threadIdx.x;            // 640*512 threads
  int c = i & (CCH - 1);
  int o = i >> 9;
  float w;
  if (o < 64)       w = Wq[o * CCH + c];
  else if (o < 128) w = Wk[(o - 64) * CCH + c];
  else              w = Wv[(o - 128) * CCH + c];
  Wqkv[i] = f2bf(w);
  if (c == 0)
    bqkv[o] = (o < 64) ? bq[o] : (o < 128) ? bk[o - 64] : bv[o - 128];
}

// ---------------- kernel 2: x (B,C,H,W) f32 -> xT (B,HW,C) bf16 ----------------
// LDS-tiled 32x32 transpose: coalesced f32 reads AND coalesced bf16 writes.
__global__ void k_xpose(const float* __restrict__ x, unsigned short* __restrict__ xT) {
  __shared__ unsigned short tile[32][34];            // pad -> conflict-free
  int n0 = blockIdx.x * 32;                          // spatial tile origin
  int c0 = blockIdx.y * 32;                          // channel tile origin
  int b  = blockIdx.z;
  int tx = threadIdx.x;                              // 0..31
  int ty = threadIdx.y;                              // 0..7
  const float* xb = x + (size_t)b * CCH * HWXY;
#pragma unroll
  for (int i = 0; i < 4; ++i) {
    int cl = ty + i * 8;
    tile[cl][tx] = f2bf(xb[(size_t)(c0 + cl) * HWXY + n0 + tx]);
  }
  __syncthreads();
  unsigned short* xo = xT + (size_t)b * HWXY * CCH;
#pragma unroll
  for (int i = 0; i < 4; ++i) {
    int nl = ty + i * 8;
    xo[(size_t)(n0 + nl) * CCH + c0 + tx] = tile[tx][nl];
  }
}

// ---------------- kernel 3: fused QKV projection GEMM ----------------
// out[n,o] = sum_c xT[n,c] * Wqkv[o,c] + bqkv[o]
// Register-blocked: each wave computes 16(n) x 64(o), A reused across 4 B tiles.
// grid (72 n-blocks of 128, 10 o-blocks of 64, B); block 256 = 8 waves (n-direction)
__global__ void k_qkv(const unsigned short* __restrict__ xT,
                      const unsigned short* __restrict__ Wqkv,
                      const float* __restrict__ bqkv,
                      unsigned short* __restrict__ qT, unsigned short* __restrict__ kT,
                      unsigned short* __restrict__ vW, unsigned short* __restrict__ vH) {
  int b    = blockIdx.z;
  int wave = threadIdx.x >> 5;
  int lane = threadIdx.x & 31;
  int nb = blockIdx.x * 128 + wave * 16;
  int ob = blockIdx.y * 64;
  const unsigned short* xb = xT + (size_t)b * HWXY * CCH;
  v8f acc0 = {}, acc1 = {}, acc2 = {}, acc3 = {};
  for (int c0 = 0; c0 < CCH; c0 += 32) {
    const unsigned short* pa = xb   + (size_t)nb * CCH + c0;
    const unsigned short* pb = Wqkv + (size_t)ob * CCH + c0;
    if (c0 + 32 < CCH) {                             // global_prefetch_b8 (near)
      __builtin_prefetch(pa + 32, 0, 3);
      __builtin_prefetch(pb + 32, 0, 3);
    }
    v16bf a  = load_a(pa, CCH, lane);
    v16bf b0 = load_b(pb,                  CCH, lane);
    v16bf b1 = load_b(pb + (size_t)16*CCH, CCH, lane);
    v16bf b2 = load_b(pb + (size_t)32*CCH, CCH, lane);
    v16bf b3 = load_b(pb + (size_t)48*CCH, CCH, lane);
    acc0 = wmma_bf16(a, b0, acc0);
    acc1 = wmma_bf16(a, b1, acc1);
    acc2 = wmma_bf16(a, b2, acc2);
    acc3 = wmma_bf16(a, b3, acc3);
  }
  int rb = (lane >> 4) * 8;
  v8f accs[4] = {acc0, acc1, acc2, acc3};
#pragma unroll
  for (int j = 0; j < 4; ++j) {
    int o = ob + j * 16 + (lane & 15);
    float bias = bqkv[o];
#pragma unroll
    for (int r = 0; r < 8; ++r) {
      int n = nb + rb + r;
      unsigned short bvv = f2bf(accs[j][r] + bias);
      if (o < 64) {
        qT[(size_t)b * HWXY * CQKD + (size_t)n * CQKD + o] = bvv;
      } else if (o < 128) {
        kT[(size_t)b * HWXY * CQKD + (size_t)n * CQKD + (o - 64)] = bvv;
      } else {
        int c = o - 128;
        int hh = n / WD, ww = n % WD;
        vW[(size_t)b * HWXY * CCH + (size_t)c * HWXY + n] = bvv;               // (C,H,W)
        vH[(size_t)b * HWXY * CCH + ((size_t)c * WD + ww) * HD + hh] = bvv;    // (C,W,H)
      }
    }
  }
}

// ---------------- kernel 4a: eH[h,g] per (b,w) column ----------------
// eH[b,h,w,g] = sum_c q[n=h*W+w,c] * k[n=g*W+w,c]  -> scores[...,0:96]
// block = 192 (6 waves); wave = h-tile; 6 accumulators over g-tiles (A reused 6x)
__global__ void k_scores_H(const unsigned short* __restrict__ qT,
                           const unsigned short* __restrict__ kT,
                           float* __restrict__ scores) {
  int w = blockIdx.x, b = blockIdx.y;
  int ht = threadIdx.x >> 5, lane = threadIdx.x & 31;
  const unsigned short* q = qT + (size_t)b * HWXY * CQKD;
  const unsigned short* k = kT + (size_t)b * HWXY * CQKD;
  float* sc = scores + (size_t)b * HWXY * 192;
  v8f acc[6] = {};
  for (int c0 = 0; c0 < CQKD; c0 += 32) {
    v16bf a = load_a(q + ((size_t)(ht * 16) * WD + w) * CQKD + c0, (size_t)WD * CQKD, lane);
#pragma unroll
    for (int gt = 0; gt < 6; ++gt) {
      v16bf bm = load_b(k + ((size_t)(gt * 16) * WD + w) * CQKD + c0, (size_t)WD * CQKD, lane);
      acc[gt] = wmma_bf16(a, bm, acc[gt]);
    }
  }
  int rb = (lane >> 4) * 8;
#pragma unroll
  for (int gt = 0; gt < 6; ++gt) {
    int g = gt * 16 + (lane & 15);
#pragma unroll
    for (int r = 0; r < 8; ++r) {
      int h = ht * 16 + rb + r;
      sc[((size_t)h * WD + w) * 192 + g] = acc[gt][r];
    }
  }
}

// ---------------- kernel 4b: eW[w,v] per (b,h) row ----------------
// eW[b,h,w,v] = sum_c q[n=h*W+w,c] * k[n=h*W+v,c]  -> scores[...,96:192]
__global__ void k_scores_W(const unsigned short* __restrict__ qT,
                           const unsigned short* __restrict__ kT,
                           float* __restrict__ scores) {
  int h = blockIdx.x, b = blockIdx.y;
  int wt = threadIdx.x >> 5, lane = threadIdx.x & 31;
  const unsigned short* q = qT + (size_t)b * HWXY * CQKD;
  const unsigned short* k = kT + (size_t)b * HWXY * CQKD;
  float* sc = scores + (size_t)b * HWXY * 192;
  v8f acc[6] = {};
  for (int c0 = 0; c0 < CQKD; c0 += 32) {
    v16bf a = load_a(q + ((size_t)h * WD + wt * 16) * CQKD + c0, CQKD, lane);
#pragma unroll
    for (int vt = 0; vt < 6; ++vt) {
      v16bf bm = load_b(k + ((size_t)h * WD + vt * 16) * CQKD + c0, CQKD, lane);
      acc[vt] = wmma_bf16(a, bm, acc[vt]);
    }
  }
  int rb = (lane >> 4) * 8;
#pragma unroll
  for (int vt = 0; vt < 6; ++vt) {
    int v = vt * 16 + (lane & 15);
#pragma unroll
    for (int r = 0; r < 8; ++r) {
      int wq = wt * 16 + rb + r;
      sc[((size_t)h * WD + wq) * 192 + 96 + v] = acc[vt][r];
    }
  }
}

// ---------------- kernel 5: softmax over 192 (diag-masked), f32 -> bf16 ----------------
// one wave32 per query pixel; 6 values/lane; shuffle reductions
__global__ void k_softmax(const float* __restrict__ scores, unsigned short* __restrict__ att) {
  int pix  = blockIdx.x * 8 + (threadIdx.x >> 5);    // grid = NTOT/8
  int lane = threadIdx.x & 31;
  int n = pix % HWXY;
  int h = n / WD;
  const float* s = scores + (size_t)pix * 192;
  float v[6];
  float m = -__builtin_inff();
#pragma unroll
  for (int i = 0; i < 6; ++i) {
    int j = lane + 32 * i;
    float x = s[j];
    if (j < 96 && j == h) x = -__builtin_inff();     // diag mask on eH
    v[i] = x;
    m = fmaxf(m, x);
  }
  for (int d = 16; d; d >>= 1) m = fmaxf(m, __shfl_xor(m, d, 32));
  float sum = 0.f;
#pragma unroll
  for (int i = 0; i < 6; ++i) { v[i] = __expf(v[i] - m); sum += v[i]; }
  for (int d = 16; d; d >>= 1) sum += __shfl_xor(sum, d, 32);
  float inv = 1.0f / sum;
  unsigned short* o = att + (size_t)pix * 192;
#pragma unroll
  for (int i = 0; i < 6; ++i) o[lane + 32 * i] = f2bf(v[i] * inv);
}

// ---------------- kernel 6a: outH -> d_out ----------------
// out[b,c,h,w] = sum_g attH[h,w,g] * v[c,g,w]; grid (W, B, 6 h-tiles)
// block 256 = 8 waves; wave computes 16(h) x 64(c), A reused across 4 B tiles.
__global__ void k_outH(const unsigned short* __restrict__ att,
                       const unsigned short* __restrict__ vH,
                       float* __restrict__ out) {
  int w = blockIdx.x, b = blockIdx.y, ht = blockIdx.z;
  int cg = threadIdx.x >> 5, lane = threadIdx.x & 31;  // cg: group of 4 c-tiles
  const unsigned short* at = att + (size_t)b * HWXY * 192;
  const unsigned short* vv = vH  + (size_t)b * HWXY * CCH;   // (C,W,H)
  float* ob = out + (size_t)b * CCH * HWXY;
  v8f acc[4] = {};
  for (int g0 = 0; g0 < HD; g0 += 32) {
    v16bf a = load_a(at + ((size_t)(ht * 16) * WD + w) * 192 + g0, (size_t)WD * 192, lane);
#pragma unroll
    for (int j = 0; j < 4; ++j) {
      int ct = cg * 4 + j;
      v16bf bm = load_b(vv + ((size_t)(ct * 16) * WD + w) * HD + g0, (size_t)WD * HD, lane);
      acc[j] = wmma_bf16(a, bm, acc[j]);
    }
  }
  int rb = (lane >> 4) * 8;
#pragma unroll
  for (int j = 0; j < 4; ++j) {
    int c = (cg * 4 + j) * 16 + (lane & 15);
#pragma unroll
    for (int r = 0; r < 8; ++r) {
      int h = ht * 16 + rb + r;
      ob[(size_t)c * HWXY + (size_t)h * WD + w] = acc[j][r];
    }
  }
}

// ---------------- kernel 6b: outW + final blend ----------------
// out = gamma * (outH + outW) + x; grid (H, B, 6 w-tiles)
__global__ void k_outW_final(const unsigned short* __restrict__ att,
                             const unsigned short* __restrict__ vW,
                             const float* __restrict__ x,
                             const float* __restrict__ gamma,
                             float* __restrict__ out) {
  int h = blockIdx.x, b = blockIdx.y, wt = blockIdx.z;
  int cg = threadIdx.x >> 5, lane = threadIdx.x & 31;
  float g = gamma[0];
  const unsigned short* at = att + (size_t)b * HWXY * 192;
  const unsigned short* vv = vW  + (size_t)b * HWXY * CCH;   // (C,H,W)
  const float* xb = x + (size_t)b * CCH * HWXY;
  float* ob = out + (size_t)b * CCH * HWXY;
  v8f acc[4] = {};
  for (int v0 = 0; v0 < WD; v0 += 32) {
    v16bf a = load_a(at + ((size_t)h * WD + wt * 16) * 192 + 96 + v0, 192, lane);
#pragma unroll
    for (int j = 0; j < 4; ++j) {
      int ct = cg * 4 + j;
      v16bf bm = load_b(vv + ((size_t)(ct * 16) * HD + h) * WD + v0, (size_t)HD * WD, lane);
      acc[j] = wmma_bf16(a, bm, acc[j]);
    }
  }
  int rb = (lane >> 4) * 8;
#pragma unroll
  for (int j = 0; j < 4; ++j) {
    int c = (cg * 4 + j) * 16 + (lane & 15);
#pragma unroll
    for (int r = 0; r < 8; ++r) {
      int w = wt * 16 + rb + r;
      size_t idx = (size_t)c * HWXY + (size_t)h * WD + w;
      ob[idx] = g * (ob[idx] + acc[j][r]) + xb[idx];
    }
  }
}

extern "C" void kernel_launch(void* const* d_in, const int* in_sizes, int n_in,
                              void* d_out, int out_size, void* d_ws, size_t ws_size,
                              hipStream_t stream) {
  const float* x     = (const float*)d_in[0];
  const float* Wq    = (const float*)d_in[1];
  const float* bq    = (const float*)d_in[2];
  const float* Wk    = (const float*)d_in[3];
  const float* bk    = (const float*)d_in[4];
  const float* Wv    = (const float*)d_in[5];
  const float* bv    = (const float*)d_in[6];
  const float* gamma = (const float*)d_in[7];
  float* out = (float*)d_out;

  unsigned char* ws = (unsigned char*)d_ws;
  size_t off = 0;
  auto take = [&](size_t bytes) -> void* {
    void* p = ws + off;
    off += (bytes + 255) & ~(size_t)255;
    return p;
  };
  unsigned short* xT    = (unsigned short*)take((size_t)NTOT * CCH * 2);   //  75.5 MB
  unsigned short* qT    = (unsigned short*)take((size_t)NTOT * CQKD * 2);  //   9.4 MB
  unsigned short* kT    = (unsigned short*)take((size_t)NTOT * CQKD * 2);  //   9.4 MB
  unsigned short* vWb   = (unsigned short*)take((size_t)NTOT * CCH * 2);   //  75.5 MB
  unsigned short* vHb   = (unsigned short*)take((size_t)NTOT * CCH * 2);   //  75.5 MB
  float*          score = (float*)         take((size_t)NTOT * 192 * 4);   //  56.6 MB
  unsigned short* att   = (unsigned short*)take((size_t)NTOT * 192 * 2);   //  28.3 MB
  unsigned short* Wqkv  = (unsigned short*)take((size_t)OQKV * CCH * 2);
  float*          bqkv  = (float*)         take((size_t)OQKV * 4);

  k_pack<<<dim3((OQKV * CCH) / 256), dim3(256), 0, stream>>>(Wq, bq, Wk, bk, Wv, bv, Wqkv, bqkv);
  k_xpose<<<dim3(HWXY / 32, CCH / 32, BB), dim3(32, 8), 0, stream>>>(x, xT);
  k_qkv<<<dim3(HWXY / 128, OQKV / 64, BB), dim3(256), 0, stream>>>(xT, Wqkv, bqkv, qT, kT, vWb, vHb);
  k_scores_H<<<dim3(WD, BB), dim3(192), 0, stream>>>(qT, kT, score);
  k_scores_W<<<dim3(HD, BB), dim3(192), 0, stream>>>(qT, kT, score);
  k_softmax<<<dim3(NTOT / 8), dim3(256), 0, stream>>>(score, att);
  k_outH<<<dim3(WD, BB, HD / 16), dim3(256), 0, stream>>>(att, vHb, out);
  k_outW_final<<<dim3(HD, BB, WD / 16), dim3(256), 0, stream>>>(att, vWb, x, gamma, out);
}
